// DynamicConvFusion_609885356647
// MI455X (gfx1250) — matrix operand order repaired and checked by reading, not compiled, not run on hardware
//
#include <hip/hip_runtime.h>
#include <hip/hip_bf16.h>
#include <math.h>

typedef _Float16 v8h  __attribute__((ext_vector_type(8)));
typedef _Float16 v16h __attribute__((ext_vector_type(16)));
typedef float    v8f  __attribute__((ext_vector_type(8)));
typedef float    v16f __attribute__((ext_vector_type(16)));

#define HW    16384      // 128*128
#define CDIM  256
#define KDIM  512
#define WTILE 128        // hw positions per block in GEMM pass
#define LDSKF 36         // padded K stride (floats) in LDS X tile (144B, 16B-aligned)

// ---------------------------------------------------------------------------
// Kernel 0: prep — convert w_fe to fp16, fold BN into scale/shift, zero pool
// ---------------------------------------------------------------------------
__global__ __launch_bounds__(256) void k_prep(
    const float* __restrict__ wfe, const float* __restrict__ gamma,
    const float* __restrict__ beta, const float* __restrict__ mean,
    const float* __restrict__ var,
    _Float16* __restrict__ wfe16, float* __restrict__ scaleArr,
    float* __restrict__ shiftArr, float* __restrict__ gpool)
{
    int tid = blockIdx.x * 256 + threadIdx.x;
    if (tid < CDIM * KDIM) wfe16[tid] = (_Float16)wfe[tid];
    if (tid < CDIM) {
        float sc = gamma[tid] / sqrtf(var[tid] + 1e-5f);
        scaleArr[tid] = sc;
        shiftArr[tid] = beta[tid] - mean[tid] * sc;
    }
    if (tid < 8 * CDIM) gpool[tid] = 0.f;   // must re-zero every launch
}

// ---------------------------------------------------------------------------
// Kernel 1: fused 1x1 conv (WMMA fp16 GEMM) + BN + ReLU + global-avg-pool
//   Double-buffered ASYNC global->LDS staging (ASYNCcnt) of the X tile,
//   transposed to [hw][k] fp32; fp16 conversion happens at A-fragment load.
//   Wave w owns hw rows [16w,16w+16): M=hw(16), N=c_out(16 x 16 tiles), K=ci.
// ---------------------------------------------------------------------------
__global__ __launch_bounds__(256) void k_gemm_pool(
    const float* __restrict__ x1, const float* __restrict__ x2,
    const _Float16* __restrict__ wfe16, const float* __restrict__ scaleArr,
    const float* __restrict__ shiftArr, float* __restrict__ gpool)
{
    __shared__ float ldsX[2 * WTILE * LDSKF];   // double-buffered [hw][k] fp32
    __shared__ float ldsPool[CDIM];

    const int tid  = threadIdx.x;
    const int b    = blockIdx.x >> 7;               // 128 blocks per batch
    const int hw0  = (blockIdx.x & 127) * WTILE;
    const int lane = tid & 31;
    const int wave = tid >> 5;
    const int row  = lane & 15;                     // M-row (A) / N-col (B,D)
    const int khalf = lane >> 4;

    ldsPool[tid] = 0.f;

    // LDS byte offsets of the two staging buffers (flat addr low 32 bits ==
    // wave-relative LDS byte address on gfx1250).
    const unsigned ldsBase[2] = {
        (unsigned)(uintptr_t)&ldsX[0],
        (unsigned)(uintptr_t)&ldsX[WTILE * LDSKF]
    };
    const float* xsrc = (const float*)((size_t)b * CDIM * HW * 4);

    // stage: issue 16 async b32 scatter-copies per wave (4096 dwords/block)
    auto stage = [&](int kc2, int buf) {
        const float* src = (kc2 < 8 ? x1 : x2)
                         + (size_t)b * CDIM * HW
                         + (size_t)((kc2 & 7) * 32) * HW + hw0;
        const unsigned lb = ldsBase[buf];
        #pragma unroll
        for (int j = 0; j < 16; ++j) {
            int seg  = wave * 16 + j;               // 0..127
            int ci   = seg >> 2;                    // 0..31
            int hsub = (seg & 3) * 32 + lane;       // 0..127 (coalesced 128B)
            const float* gp = src + (size_t)ci * HW + hsub;
            unsigned loff = lb + (unsigned)((hsub * LDSKF + ci) * 4);
            asm volatile("global_load_async_to_lds_b32 %0, %1, off"
                         :: "v"(loff), "v"(gp) : "memory");
        }
    };

    v8f acc[16] = {};                               // 16 N-tiles of 16x16 f32

    stage(0, 0);

    for (int kc = 0; kc < 16; ++kc) {               // K chunks of 32
        const int cur = kc & 1;
        if (kc < 15) stage(kc + 1, cur ^ 1);
        // wait for the CURRENT buffer's 16 asyncs (next buffer's may remain)
        if (kc < 15) asm volatile("s_wait_asynccnt 0x10" ::: "memory");
        else         asm volatile("s_wait_asynccnt 0x0"  ::: "memory");
        __syncthreads();

        // ---- A fragment (X tile): 16-bit A layout: lanes<16 K={0..7,16..23},
        //      lanes>=16 K={8..15,24..31}; fp32 LDS loads + cvt to fp16.
        const float* ap = &ldsX[(size_t)cur * WTILE * LDSKF
                                + (wave * 16 + row) * LDSKF + khalf * 8];
        v8f alo = *(const v8f*)ap;
        v8f ahi = *(const v8f*)(ap + 16);
        v16f af = __builtin_shufflevector(alo, ahi,
            0,1,2,3,4,5,6,7,8,9,10,11,12,13,14,15);
        v16h afrag = __builtin_convertvector(af, v16h);

        const int kbase = kc * 32 + khalf * 8;      // B: lanes<16 K={0..7,16..23}
        const _Float16* wrow = wfe16 + kbase;

        // distance-1 software pipeline on the B-fragment loads
        v8h blo = *(const v8h*)(wrow + (size_t)row * KDIM);
        v8h bhi = *(const v8h*)(wrow + (size_t)row * KDIM + 16);
        #pragma unroll
        for (int nt = 0; nt < 16; ++nt) {
            v8h nlo = blo, nhi = bhi;
            if (nt < 15) {
                const _Float16* bp = wrow + (size_t)((nt + 1) * 16 + row) * KDIM;
                nlo = *(const v8h*)bp;
                nhi = *(const v8h*)(bp + 16);
            }
            v16h bfrag = __builtin_shufflevector(blo, bhi,
                0,1,2,3,4,5,6,7,8,9,10,11,12,13,14,15);
            acc[nt] = __builtin_amdgcn_wmma_f32_16x16x32_f16(
                false, afrag, false, bfrag, (short)0, acc[nt], false, false);
            blo = nlo; bhi = nhi;
        }
        __syncthreads();   // all waves done reading buf 'cur' before restage
    }
    (void)xsrc;

    // ---- epilogue: BN + ReLU + reduce over the 16 hw rows of each tile
    #pragma unroll
    for (int nt = 0; nt < 16; ++nt) {
        const int c = nt * 16 + row;                // N = lane&15
        const float sc = scaleArr[c];
        const float sh = shiftArr[c];
        float s = 0.f;
        #pragma unroll
        for (int r = 0; r < 8; ++r) {               // 8 M-rows per lane-half
            float v = acc[nt][r] * sc + sh;
            s += (v > 0.f) ? v : 0.f;
        }
        s += __shfl_xor(s, 16, 32);                 // combine M 0-7 with 8-15
        if (lane < 16) atomicAdd(&ldsPool[c], s);   // ds_add_f32
    }
    __syncthreads();
    atomicAdd(&gpool[b * CDIM + tid], ldsPool[tid]); // 256 atomics per block
}

// ---------------------------------------------------------------------------
// Kernel 2: tiny MLPs — pooled -> dynamic 3x3 weights [B*C,9] and attn [B,2]
// ---------------------------------------------------------------------------
__global__ __launch_bounds__(256) void k_mlp(
    const float* __restrict__ gpool,
    const float* __restrict__ w_cg1, const float* __restrict__ w_cg2,
    const float* __restrict__ w_ag1, const float* __restrict__ w_ag2,
    float* __restrict__ cw, float* __restrict__ attnOut)
{
    __shared__ float pooled[8 * CDIM];
    __shared__ float hcg[8 * 64];
    __shared__ float hag[8 * 64];
    const int tid = threadIdx.x;
    const float invHW = 1.f / (float)HW;

    for (int i = tid; i < 8 * CDIM; i += 256) pooled[i] = gpool[i] * invHW;
    __syncthreads();

    for (int o = tid; o < 8 * 64; o += 256) {
        int bb = o >> 6, j = o & 63;
        const float* p  = &pooled[bb * CDIM];
        const float* w1 = &w_cg1[j * CDIM];
        const float* w2 = &w_ag1[j * CDIM];
        float s1 = 0.f, s2 = 0.f;
        for (int i = 0; i < CDIM; ++i) { s1 += p[i] * w1[i]; s2 += p[i] * w2[i]; }
        hcg[o] = fmaxf(s1, 0.f);
        hag[o] = fmaxf(s2, 0.f);
    }
    __syncthreads();

    for (int o = tid; o < 8 * 9 * CDIM; o += 256) {   // 18432 outputs
        int bb = o / (9 * CDIM), r = o % (9 * CDIM);  // r = c*9 + k
        const float* h = &hcg[bb * 64];
        const float* w = &w_cg2[r * 64];
        float s = 0.f;
        for (int j = 0; j < 64; ++j) s += h[j] * w[j];
        cw[o] = s;                                    // layout = (b*256+c)*9+k
    }
    if (tid < 16) {
        int bb = tid >> 1, k = tid & 1;
        const float* h = &hag[bb * 64];
        const float* w = &w_ag2[k * 64];
        float s = 0.f;
        for (int j = 0; j < 64; ++j) s += h[j] * w[j];
        attnOut[tid] = 1.f / (1.f + __expf(-s));
    }
}

// ---------------------------------------------------------------------------
// Kernel 3: depthwise 3x3 (cross-correlation, SAME) on x1,x2 + combine
//   out = 0.5*(x1+x2) + 0.25*a1*d1 + 0.25*a2*d2   (LAMBDA_S=LAMBDA_C=0.5)
// ---------------------------------------------------------------------------
#define TROWS 16
__global__ __launch_bounds__(256) void k_dwconv(
    const float* __restrict__ x1, const float* __restrict__ x2,
    const float* __restrict__ cw, const float* __restrict__ attn,
    float* __restrict__ out)
{
    __shared__ float t1[(TROWS + 2) * 130];
    __shared__ float t2[(TROWS + 2) * 130];
    const int tid = threadIdx.x;
    const int blk = blockIdx.x;            // b*2048 + c*8 + rowtile
    const int rt  = blk & 7;
    const int c   = (blk >> 3) & 255;
    const int b   = blk >> 11;
    const int r0  = rt * TROWS;
    const size_t plane = (size_t)(b * CDIM + c) * HW;
    const float* p1 = x1 + plane;
    const float* p2 = x2 + plane;

    for (int idx = tid; idx < (TROWS + 2) * 130; idx += 256) {
        int rr = idx / 130, cc = idx % 130;
        int gr = r0 + rr - 1, gc = cc - 1;
        bool in = (gr >= 0) & (gr < 128) & (gc >= 0) & (gc < 128);
        float a = 0.f, bv = 0.f;
        if (in) { a = p1[gr * 128 + gc]; bv = p2[gr * 128 + gc]; }
        t1[idx] = a;
        t2[idx] = bv;
    }
    float w[9];
    const float* wp = &cw[(size_t)(b * CDIM + c) * 9];
    #pragma unroll
    for (int k = 0; k < 9; ++k) w[k] = wp[k];
    const float a1 = attn[b * 2 + 0] * 0.25f;
    const float a2 = attn[b * 2 + 1] * 0.25f;
    __syncthreads();

    #pragma unroll
    for (int j = 0; j < 8; ++j) {
        int pix = j * 256 + tid;           // 2048 px = 16 rows x 128 cols
        int pr = pix >> 7, pc = pix & 127;
        float d1 = 0.f, d2 = 0.f;
        #pragma unroll
        for (int dy = 0; dy < 3; ++dy)
            #pragma unroll
            for (int dx = 0; dx < 3; ++dx) {
                float wk = w[dy * 3 + dx];
                d1 += t1[(pr + dy) * 130 + pc + dx] * wk;
                d2 += t2[(pr + dy) * 130 + pc + dx] * wk;
            }
        float v1 = t1[(pr + 1) * 130 + pc + 1];
        float v2 = t2[(pr + 1) * 130 + pc + 1];
        out[plane + (size_t)(r0 + pr) * 128 + pc]
            = 0.5f * (v1 + v2) + a1 * d1 + a2 * d2;
    }
}

// ---------------------------------------------------------------------------
// Launch
// ---------------------------------------------------------------------------
extern "C" void kernel_launch(void* const* d_in, const int* in_sizes, int n_in,
                              void* d_out, int out_size, void* d_ws, size_t ws_size,
                              hipStream_t stream)
{
    const float* x1    = (const float*)d_in[0];
    const float* x2    = (const float*)d_in[1];
    const float* w_fe  = (const float*)d_in[2];
    const float* bn_g  = (const float*)d_in[3];
    const float* bn_b  = (const float*)d_in[4];
    const float* bn_m  = (const float*)d_in[5];
    const float* bn_v  = (const float*)d_in[6];
    const float* w_cg1 = (const float*)d_in[7];
    const float* w_cg2 = (const float*)d_in[8];
    const float* w_ag1 = (const float*)d_in[9];
    const float* w_ag2 = (const float*)d_in[10];
    float* outp = (float*)d_out;

    // workspace layout (bytes)
    char* ws = (char*)d_ws;
    _Float16* wfe16    = (_Float16*)(ws);                   // 256*512*2 = 262144
    float*    scaleArr = (float*)(ws + 262144);             // 1024
    float*    shiftArr = (float*)(ws + 263168);             // 1024
    float*    gpool    = (float*)(ws + 264192);             // 8192
    float*    cw       = (float*)(ws + 272384);             // 73728
    float*    attn     = (float*)(ws + 346112);             // 64

    k_prep<<<512, 256, 0, stream>>>(w_fe, bn_g, bn_b, bn_m, bn_v,
                                    wfe16, scaleArr, shiftArr, gpool);
    k_gemm_pool<<<8 * 128, 256, 0, stream>>>(x1, x2, wfe16, scaleArr,
                                             shiftArr, gpool);
    k_mlp<<<1, 256, 0, stream>>>(gpool, w_cg1, w_cg2, w_ag1, w_ag2, cw, attn);
    k_dwconv<<<8 * 256 * 8, 256, 0, stream>>>(x1, x2, cw, attn, outp);
}
